// UniversalCalculator_57114475102483
// MI455X (gfx1250) — compile-verified, matrix operand order checked by reading, not run on hardware
//
#include <hip/hip_runtime.h>
#include <cstdint>

// ---------------- problem constants ----------------
#define T_TOK 8192
#define DDIM  2048
#define HDIM  1408
#define NEXP  8
#define KTOP  2
#define BM    64
#define KB    32
#define MAXR  (T_TOK * KTOP + NEXP * BM)   // 16896 padded assignment rows
#define MT    (MAXR / BM)                  // 264 max M tiles

typedef __attribute__((ext_vector_type(16))) __bf16 v16bf;
typedef __attribute__((ext_vector_type(2)))  __bf16 v2bf;
typedef __attribute__((ext_vector_type(8)))  float  v8f;

__device__ __forceinline__ v8f wmma_bf16(v16bf a, v16bf b, v8f c) {
  // D = A(16x32 bf16) * B(32x16 bf16) + C(16x16 f32)
  return __builtin_amdgcn_wmma_f32_16x16x32_bf16(false, a, false, b, (short)0, c,
                                                 false, false);
}

// two fp32 -> packed bf16 pair; native fptrunc-to-bf16 lets ISel pick
// v_cvt_pk_bf16_f32 on targets that have it.
__device__ __forceinline__ unsigned pack_bf16(float lo, float hi) {
#if __has_builtin(__builtin_amdgcn_cvt_pk_bf16_f32)
  v2bf p = __builtin_amdgcn_cvt_pk_bf16_f32(lo, hi);
#else
  v2bf p = {(__bf16)lo, (__bf16)hi};
#endif
  return __builtin_bit_cast(unsigned, p);
}

__device__ __forceinline__ float silu_f(float z) {
  return z / (1.0f + __expf(-z));
}

// gfx1250 async global->LDS copy (16B per lane, tracked by ASYNCcnt).
__device__ __forceinline__ void async_load_b128(unsigned lds_off, const void* gptr) {
  asm volatile("global_load_async_to_lds_b128 %0, %1, off"
               :: "v"(lds_off), "v"(gptr) : "memory");
}
__device__ __forceinline__ void wait_async() {
  asm volatile("s_wait_asynccnt 0x0" ::: "memory");
}
__device__ __forceinline__ unsigned lds_addr32(const void* p) {
  return (unsigned)(uintptr_t)p;   // LDS aperture: low 32 bits = LDS offset
}

// ---------------- router ----------------
__global__ void moe_init(int* counts, int* rowTok, float* rowW) {
  int g = blockIdx.x * blockDim.x + threadIdx.x;
  if (g < 16) counts[g] = 0;                 // counts[8] + cursors[8]
  int stride = gridDim.x * blockDim.x;
  for (int i = g; i < MAXR; i += stride) { rowTok[i] = -1; rowW[i] = 0.0f; }
}

__global__ void moe_count(const int* __restrict__ idxs, int* counts, int n) {
  int i = blockIdx.x * blockDim.x + threadIdx.x;
  if (i < n) atomicAdd(&counts[idxs[i] & (NEXP - 1)], 1);
}

__global__ void moe_offsets(const int* __restrict__ counts, int* offsets) {
  if (threadIdx.x == 0 && blockIdx.x == 0) {
    int acc = 0;
    offsets[0] = 0;
    for (int e = 0; e < NEXP; ++e) {
      acc += (counts[e] + BM - 1) & ~(BM - 1);   // BM-aligned segments
      offsets[e + 1] = acc;
    }
  }
}

__global__ void moe_scatter(const int* __restrict__ idxs,
                            const float* __restrict__ scores,
                            const int* __restrict__ offsets, int* cursors,
                            int* rowTok, float* rowW, int n) {
  int i = blockIdx.x * blockDim.x + threadIdx.x;
  if (i < n) {
    int e = idxs[i] & (NEXP - 1);
    int p = offsets[e] + atomicAdd(&cursors[e], 1);
    rowTok[p] = i >> 1;                         // KTOP == 2
    rowW[p]   = scores[i];
  }
}

// ---------------- x -> bf16 ----------------
__global__ void cvt_x(const float* __restrict__ x, unsigned short* __restrict__ xbf) {
  size_t i = ((size_t)blockIdx.x * blockDim.x + threadIdx.x) * 8;
  if (i + 8 <= (size_t)T_TOK * DDIM) {
    const float4* s = (const float4*)(x + i);
    float4 a = s[0], b = s[1];
    uint4 o;
    o.x = pack_bf16(a.x, a.y);
    o.y = pack_bf16(a.z, a.w);
    o.z = pack_bf16(b.x, b.y);
    o.w = pack_bf16(b.z, b.w);
    *(uint4*)(xbf + i) = o;
  }
}

// ---------------- GEMM1: h = silu(x*Wg) * (x*Wu), bf16 out ----------------
// grid: (MT, HDIM/64), 256 threads (8 waves); waves 0-3 gate, 4-7 up.
// Double-buffered LDS; A tile via async global->LDS, B tiles via VGPR cvt.
__global__ __launch_bounds__(256) void moe_gate_up(
    const unsigned short* __restrict__ xbf,
    const float* __restrict__ Wg, const float* __restrict__ Wu,
    const int* __restrict__ offsets, const int* __restrict__ rowTok,
    unsigned short* __restrict__ Hbuf) {
  int m0 = blockIdx.x * BM;
  if (m0 >= offsets[NEXP]) return;
  int e = 0;
  while (m0 >= offsets[e + 1]) ++e;           // tiles never straddle experts
  int n0 = blockIdx.y * 64;
  const float* wg = Wg + (size_t)e * DDIM * HDIM;
  const float* wu = Wu + (size_t)e * DDIM * HDIM;

  __shared__ __align__(32) unsigned short lsA[2][BM * KB];   // fragment order
  __shared__ __align__(32) unsigned short lsBg[2][KB * 64];
  __shared__ __align__(32) unsigned short lsBu[2][KB * 64];
  __shared__ float xch[BM * 64];
  __shared__ int sTok[BM];

  int tid = threadIdx.x;
  int wave = tid >> 5, lane = tid & 31;
  if (tid < BM) sTok[tid] = rowTok[m0 + tid];
  __syncthreads();                            // sTok ready for prologue

  bool isUp = wave >= 4;
  int ms = wave & 3;

  v8f zero = {0.f, 0.f, 0.f, 0.f, 0.f, 0.f, 0.f, 0.f};
  v8f acc[4];
#pragma unroll
  for (int i = 0; i < 4; ++i) acc[i] = zero;

  // A staging: thread -> (row, 8 consecutive k), WMMA-fragment swizzle
  int rA = tid >> 2, kqA = (tid & 3) * 8;
  int ldsA_off = (rA >> 4) * 512 +
                 ((rA & 15) + (((kqA >> 3) & 1) << 4)) * 16 + (kqA >> 4) * 8;
  // B staging: thread -> (k-row, 8 consecutive cols in one 16-col group)
  int kr = tid >> 3, c0 = (tid & 7) * 8;
  int eB = (kr & 7) + ((kr >> 4) << 3);
  int lhB = ((kr >> 3) & 1) << 4;
  int baseB = (c0 >> 4) * 512 + ((c0 & 15) + lhB) * 16 + eB;

  struct BReg { float4 g0, g1, u0, u1; };

  auto loadA = [&](int buf, int k0) {
    int tok = sTok[rA];
    if (tok >= 0) {
      async_load_b128(lds_addr32(&lsA[buf][ldsA_off]),
                      xbf + (size_t)tok * DDIM + k0 + kqA);
    } else {
      *(uint4*)(&lsA[buf][ldsA_off]) = make_uint4(0u, 0u, 0u, 0u);
    }
  };
  auto fetchB = [&](int k0) {
    BReg r;
    const float4* g = (const float4*)(wg + (size_t)(k0 + kr) * HDIM + n0 + c0);
    r.g0 = g[0]; r.g1 = g[1];
    const float4* u = (const float4*)(wu + (size_t)(k0 + kr) * HDIM + n0 + c0);
    r.u0 = u[0]; r.u1 = u[1];
    return r;
  };
  auto storeB = [&](int buf, const BReg& r) {
    float gv[8] = {r.g0.x, r.g0.y, r.g0.z, r.g0.w, r.g1.x, r.g1.y, r.g1.z, r.g1.w};
    float uv[8] = {r.u0.x, r.u0.y, r.u0.z, r.u0.w, r.u1.x, r.u1.y, r.u1.z, r.u1.w};
#pragma unroll
    for (int j = 0; j < 8; j += 2) {
      unsigned pg = pack_bf16(gv[j], gv[j + 1]);
      unsigned pu = pack_bf16(uv[j], uv[j + 1]);
      int off = baseB + j * 16;                 // adjacent cols: +16 shorts
      lsBg[buf][off]      = (unsigned short)pg;
      lsBg[buf][off + 16] = (unsigned short)(pg >> 16);
      lsBu[buf][off]      = (unsigned short)pu;
      lsBu[buf][off + 16] = (unsigned short)(pu >> 16);
    }
  };

  // prologue: stage tile 0
  loadA(0, 0);
  { BReg b0 = fetchB(0); storeB(0, b0); }
  wait_async();
  __syncthreads();

  int buf = 0;
  for (int k0 = 0; k0 < DDIM; k0 += KB, buf ^= 1) {
    int kn = k0 + KB;
    BReg bn = {};
    if (kn < DDIM) { loadA(buf ^ 1, kn); bn = fetchB(kn); }   // prefetch next

    const unsigned short* lsB = isUp ? lsBu[buf] : lsBg[buf];
    v16bf a  = *(const v16bf*)(lsA[buf] + ms * 512 + lane * 16);
    v16bf b0 = *(const v16bf*)(lsB + 0 * 512 + lane * 16);
    v16bf b1 = *(const v16bf*)(lsB + 1 * 512 + lane * 16);
    v16bf b2 = *(const v16bf*)(lsB + 2 * 512 + lane * 16);
    v16bf b3 = *(const v16bf*)(lsB + 3 * 512 + lane * 16);
    acc[0] = wmma_bf16(a, b0, acc[0]);
    acc[1] = wmma_bf16(a, b1, acc[1]);
    acc[2] = wmma_bf16(a, b2, acc[2]);
    acc[3] = wmma_bf16(a, b3, acc[3]);

    if (kn < DDIM) storeB(buf ^ 1, bn);
    wait_async();
    __syncthreads();
  }

  // fuse SwiGLU: up waves publish, gate waves combine in place
  if (isUp) {
#pragma unroll
    for (int ns = 0; ns < 4; ++ns)
#pragma unroll
      for (int r = 0; r < 8; ++r) {
        int mm = ms * 16 + r + ((lane >> 4) << 3);
        int nn = ns * 16 + (lane & 15);
        xch[mm * 64 + nn] = acc[ns][r];
      }
  }
  __syncthreads();
  if (!isUp) {
#pragma unroll
    for (int ns = 0; ns < 4; ++ns)
#pragma unroll
      for (int r = 0; r < 8; ++r) {
        int mm = ms * 16 + r + ((lane >> 4) << 3);
        int nn = ns * 16 + (lane & 15);
        xch[mm * 64 + nn] = silu_f(acc[ns][r]) * xch[mm * 64 + nn];
      }
  }
  __syncthreads();
  {   // coalesced bf16 store of the 64x64 h tile
    int r = tid >> 2, cc = (tid & 3) * 16;
    unsigned int pk[8];
#pragma unroll
    for (int j = 0; j < 8; ++j)
      pk[j] = pack_bf16(xch[r * 64 + cc + 2 * j], xch[r * 64 + cc + 2 * j + 1]);
    uint4* dst = (uint4*)(Hbuf + (size_t)(m0 + r) * HDIM + n0 + cc);
    dst[0] = make_uint4(pk[0], pk[1], pk[2], pk[3]);
    dst[1] = make_uint4(pk[4], pk[5], pk[6], pk[7]);
  }
}

// ---------------- GEMM2: y[tok] += w * (h * W_down) ----------------
// grid: (MT, DDIM/128), 256 threads; wave -> (ms = w&3, n-half = w>>2).
// Double-buffered; A tile (already bf16, rows contiguous) fully async.
__global__ __launch_bounds__(256) void moe_down(
    const unsigned short* __restrict__ Hbuf, const float* __restrict__ Wd,
    const int* __restrict__ offsets, const int* __restrict__ rowTok,
    const float* __restrict__ rowW, float* __restrict__ y) {
  int m0 = blockIdx.x * BM;
  if (m0 >= offsets[NEXP]) return;
  int e = 0;
  while (m0 >= offsets[e + 1]) ++e;
  int n0 = blockIdx.y * 128;
  const float* wd = Wd + (size_t)e * HDIM * DDIM;

  __shared__ __align__(32) unsigned short lsA[2][BM * KB];    // 64x32
  __shared__ __align__(32) unsigned short lsB[2][KB * 128];   // 32x128
  __shared__ int sTok[BM];
  __shared__ float sW[BM];

  int tid = threadIdx.x, wave = tid >> 5, lane = tid & 31;
  if (tid < BM) { sTok[tid] = rowTok[m0 + tid]; sW[tid] = rowW[m0 + tid]; }
  int ms = wave & 3, nh = wave >> 2;

  v8f zero = {0.f, 0.f, 0.f, 0.f, 0.f, 0.f, 0.f, 0.f};
  v8f acc[4];
#pragma unroll
  for (int i = 0; i < 4; ++i) acc[i] = zero;

  int rA = tid >> 2, kqA = (tid & 3) * 8;
  int ldsA_off = (rA >> 4) * 512 +
                 ((rA & 15) + (((kqA >> 3) & 1) << 4)) * 16 + (kqA >> 4) * 8;
  int kr = tid >> 3, c0 = (tid & 7) * 16;     // 16 cols = one whole col-group
  int eB = (kr & 7) + ((kr >> 4) << 3);
  int lhB = ((kr >> 3) & 1) << 4;
  int baseB = (c0 >> 4) * 512 + lhB * 16 + eB;   // c0&15 == 0

  struct BReg { float4 v0, v1, v2, v3; };

  auto loadA = [&](int buf, int k0) {
    async_load_b128(lds_addr32(&lsA[buf][ldsA_off]),
                    Hbuf + (size_t)(m0 + rA) * HDIM + k0 + kqA);
  };
  auto fetchB = [&](int k0) {
    const float4* g = (const float4*)(wd + (size_t)(k0 + kr) * DDIM + n0 + c0);
    BReg r; r.v0 = g[0]; r.v1 = g[1]; r.v2 = g[2]; r.v3 = g[3];
    return r;
  };
  auto storeB = [&](int buf, const BReg& r) {
    float vv[16] = {r.v0.x, r.v0.y, r.v0.z, r.v0.w, r.v1.x, r.v1.y, r.v1.z, r.v1.w,
                    r.v2.x, r.v2.y, r.v2.z, r.v2.w, r.v3.x, r.v3.y, r.v3.z, r.v3.w};
#pragma unroll
    for (int j = 0; j < 16; j += 2) {
      unsigned p = pack_bf16(vv[j], vv[j + 1]);
      int off = baseB + j * 16;
      lsB[buf][off]      = (unsigned short)p;
      lsB[buf][off + 16] = (unsigned short)(p >> 16);
    }
  };

  loadA(0, 0);
  { BReg b0 = fetchB(0); storeB(0, b0); }
  wait_async();
  __syncthreads();

  int buf = 0;
  for (int k0 = 0; k0 < HDIM; k0 += KB, buf ^= 1) {
    int kn = k0 + KB;
    BReg bn = {};
    if (kn < HDIM) { loadA(buf ^ 1, kn); bn = fetchB(kn); }

    v16bf a  = *(const v16bf*)(lsA[buf] + ms * 512 + lane * 16);
    const unsigned short* lb = lsB[buf] + (nh * 4) * 512 + lane * 16;
    v16bf b0 = *(const v16bf*)(lb + 0 * 512);
    v16bf b1 = *(const v16bf*)(lb + 1 * 512);
    v16bf b2 = *(const v16bf*)(lb + 2 * 512);
    v16bf b3 = *(const v16bf*)(lb + 3 * 512);
    acc[0] = wmma_bf16(a, b0, acc[0]);
    acc[1] = wmma_bf16(a, b1, acc[1]);
    acc[2] = wmma_bf16(a, b2, acc[2]);
    acc[3] = wmma_bf16(a, b3, acc[3]);

    if (kn < HDIM) storeB(buf ^ 1, bn);
    wait_async();
    __syncthreads();
  }

  // epilogue: scale by gate weight, scatter-add into y (K=2 collisions)
#pragma unroll
  for (int i = 0; i < 4; ++i) {
    int nn = (nh * 4 + i) * 16 + (lane & 15);
#pragma unroll
    for (int r = 0; r < 8; ++r) {
      int mm = ms * 16 + r + ((lane >> 4) << 3);
      int tok = sTok[mm];
      if (tok >= 0)
        atomicAdd(y + (size_t)tok * DDIM + n0 + nn, acc[i][r] * sW[mm]);
    }
  }
}

// ---------------- host launcher ----------------
extern "C" void kernel_launch(void* const* d_in, const int* in_sizes, int n_in,
                              void* d_out, int out_size, void* d_ws,
                              size_t ws_size, hipStream_t stream) {
  (void)in_sizes; (void)n_in; (void)out_size; (void)ws_size;
  const float* x       = (const float*)d_in[0];
  const int*   topkIdx = (const int*)d_in[1];     // harness: integer -> int32
  const float* topkW   = (const float*)d_in[2];
  const float* Wg      = (const float*)d_in[3];
  const float* Wu      = (const float*)d_in[4];
  const float* Wd      = (const float*)d_in[5];
  float* y = (float*)d_out;

  char* ws = (char*)d_ws;
  int* counts  = (int*)ws;                 // 8 counts + 8 cursors
  int* cursors = counts + 8;
  int* offsets = (int*)(ws + 64);          // 9 ints
  int* rowTok  = (int*)(ws + 128);         // MAXR
  float* rowW  = (float*)(ws + 128 + (size_t)MAXR * 4);
  size_t xbf_off = (128 + (size_t)MAXR * 8 + 255) & ~(size_t)255;
  unsigned short* xbf  = (unsigned short*)(ws + xbf_off);      // T*D bf16
  unsigned short* Hbuf = xbf + (size_t)T_TOK * DDIM;           // MAXR*H bf16

  int nAssign = T_TOK * KTOP;
  moe_init<<<64, 256, 0, stream>>>(counts, rowTok, rowW);
  moe_count<<<(nAssign + 255) / 256, 256, 0, stream>>>(topkIdx, counts, nAssign);
  moe_offsets<<<1, 32, 0, stream>>>(counts, offsets);
  moe_scatter<<<(nAssign + 255) / 256, 256, 0, stream>>>(
      topkIdx, topkW, offsets, cursors, rowTok, rowW, nAssign);
  cvt_x<<<(int)(((size_t)T_TOK * DDIM / 8 + 255) / 256), 256, 0, stream>>>(x, xbf);
  moe_gate_up<<<dim3(MT, HDIM / 64), 256, 0, stream>>>(xbf, Wg, Wu, offsets,
                                                       rowTok, Hbuf);
  hipMemsetAsync(y, 0, (size_t)T_TOK * DDIM * sizeof(float), stream);
  moe_down<<<dim3(MT, DDIM / 128), 256, 0, stream>>>(Hbuf, Wd, offsets, rowTok,
                                                     rowW, y);
}
